// QuerySAT_31679678775972
// MI455X (gfx1250) — matrix operand order repaired and checked by reading, not compiled, not run on hardware
//
#include <hip/hip_runtime.h>

// ---------------------------------------------------------------------------
// QuerySAT step network on MI455X (gfx1250), bf16 WMMA path.
//
// Input flattening assumption (jax pytree leaves, dict keys sorted):
//  0: variables_init [20000,128] f32
//  1..6  : forget_gate      W0[256,128] b0 W1[128,128] b1 W2[128,128] b2
//  7     : ln_beta[128]   8: ln_gamma[128]
//  9..14 : query_neg_inter W0[128,128] b0 W1[128,128] b1 W2[128,64] b2
// 15..20 : query_pos_inter (same shapes)
// 21..26 : update_gate     W0[256,128] b0 W1[128,128] b1 W2[128,128] b2
// 27..32 : variables_output W0[128,128] b0 W1[128,128] b1 W2[128,1] b2
// 33..38 : variables_query W0..W2 all [128,128]
// 39: lit_vars[252000] i32  40: lit_sign[252000] f32  41: clause_ids i32
// 42: n_vars  43: n_clauses
// ---------------------------------------------------------------------------

#define NVARS    20000
#define NCLAUSES 84000
#define NLITS    252000
#define NROUNDS  4

typedef __attribute__((ext_vector_type(16))) __bf16 bf16x16;
typedef __attribute__((ext_vector_type(8)))  float  floatx8;

union ABFrag { uint4 q[2]; bf16x16 v; };
static_assert(sizeof(ABFrag) == 32, "frag size");

__device__ __forceinline__ unsigned short f2bf(float f) {
    unsigned int u = __float_as_uint(f);
    unsigned int r = u + 0x7FFFu + ((u >> 16) & 1u);   // round-to-nearest-even
    return (unsigned short)(r >> 16);
}

__device__ __forceinline__ float actf(float x, int act) {
    if (act == 1) return x > 0.f ? x : 0.f;                 // relu
    if (act == 2) return 1.f / (1.f + __expf(-x));          // sigmoid
    return x;
}

// ---------------------------------------------------------------------------
// Pack an f32 [K,N] weight matrix into per-lane WMMA bf16 B-fragment order:
// element e of lane l in tile (kt,nt) is  K = kt*32 + (l>>4)*16 + e,
//                                         N = nt*16 + (l&15).
// ---------------------------------------------------------------------------
__global__ __launch_bounds__(256) void pack_weights(
    const float* __restrict__ W, unsigned short* __restrict__ out, int K, int N)
{
    int idx = blockIdx.x * 256 + threadIdx.x;
    if (idx >= K * N) return;
    int k = idx / N, n = idx % N;
    int numNt = N >> 4;
    int kt = k >> 5, rem = k & 31, hlf = rem >> 4, e = rem & 15;
    int nt = n >> 4, ln = n & 15;
    int lane = hlf * 16 + ln;
    size_t o = (((size_t)kt * numNt + nt) * 32 + lane) * 16 + e;
    out[o] = f2bf(W[(size_t)k * N + n]);
}

// ---------------------------------------------------------------------------
// One wave computes a 16(M) x (NT*16)(N) tile via v_wmma_f32_16x16x32_bf16.
// A fragments come from bf16 activations in LDS (stride in ushorts, multiple
// of 8 so uint4 loads stay 16B aligned); B fragments from pre-packed weights.
// ---------------------------------------------------------------------------
template<int KDIM, int NT>
__device__ __forceinline__ void gemm_tile(
    const unsigned short* ldsAct, int stride,
    const unsigned short* __restrict__ Wp,
    int lane, int wrb, floatx8* acc)
{
    const int half = lane >> 4, ln = lane & 15;
#pragma unroll
    for (int nt = 0; nt < NT; ++nt) {
        floatx8 z = {0.f,0.f,0.f,0.f,0.f,0.f,0.f,0.f};
        acc[nt] = z;
    }
#pragma unroll
    for (int kt = 0; kt < KDIM / 32; ++kt) {
        ABFrag a;
        const unsigned short* rp = ldsAct + (wrb + ln) * stride + kt * 32 + half * 16;
        a.q[0] = *(const uint4*)(rp);        // K = kt*32 + half*16 + [0..7]
        a.q[1] = *(const uint4*)(rp + 8);    // K = kt*32 + half*16 + 8 + [0..7]
#pragma unroll
        for (int nt = 0; nt < NT; ++nt) {
            ABFrag b;
            const uint4* wq = (const uint4*)(Wp + (((size_t)kt * NT + nt) * 32 + lane) * 16);
            b.q[0] = wq[0];
            b.q[1] = wq[1];
            acc[nt] = __builtin_amdgcn_wmma_f32_16x16x32_bf16(
                false, a.v, false, b.v, (short)0, acc[nt], false, false);
        }
    }
}

__device__ __forceinline__ void store_tile_lds(
    unsigned short* lds, int stride, int wrb, int lane,
    const floatx8* acc, const float* __restrict__ Bias, int numNt, int act)
{
    const int half = lane >> 4, ln = lane & 15;
    for (int nt = 0; nt < numNt; ++nt) {
        float bb = Bias[nt * 16 + ln];
#pragma unroll
        for (int r = 0; r < 8; ++r) {
            float v = actf(acc[nt][r] + bb, act);
            lds[(wrb + half * 8 + r) * stride + nt * 16 + ln] = f2bf(v);
        }
    }
}

__device__ __forceinline__ void store_tile_global(
    float* __restrict__ Y, int outStride, int rows, int rowBlock,
    int wrb, int lane, const floatx8* acc, const float* __restrict__ Bias,
    int numNt, int act)
{
    const int half = lane >> 4, ln = lane & 15;
    for (int nt = 0; nt < numNt; ++nt) {
        float bb = Bias[nt * 16 + ln];
#pragma unroll
        for (int r = 0; r < 8; ++r) {
            int grow = rowBlock + wrb + half * 8 + r;
            if (grow < rows) {
                float v = actf(acc[nt][r] + bb, act);
                Y[(size_t)grow * outStride + nt * 16 + ln] = v;
            }
        }
    }
}

// ---------------------------------------------------------------------------
// Fused 2/3-layer MLP: 64 rows per 128-thread block (4 waves x 16 rows).
// Hidden width fixed at 128; first layer input K0 in {128,256}; last layer
// output NOUT in {64,128}. Mid layers relu; finalAct runtime (0/1/2).
// ---------------------------------------------------------------------------
template<int K0, int NOUT, int NLAYERS>
__global__ __launch_bounds__(128) void mlp_wmma(
    const float* __restrict__ X, int inStride, int rows,
    const unsigned short* __restrict__ Wp0, const float* __restrict__ B0,
    const unsigned short* __restrict__ Wp1, const float* __restrict__ B1,
    const unsigned short* __restrict__ Wp2, const float* __restrict__ B2,
    float* __restrict__ Y, int outStride, int finalAct)
{
    constexpr int MT = 64;
    constexpr int S0 = K0 + 8;    // ushort stride, multiple of 8 -> 16B aligned rows
    constexpr int S1 = 128 + 8;
    static_assert(S1 <= S0, "layer-2 reuses ldsIn buffer");
    __shared__ __align__(16) unsigned short ldsIn[MT * S0];
    __shared__ __align__(16) unsigned short ldsMid[MT * S1];

    const int tid  = threadIdx.x;
    const int lane = tid & 31;
    const int wave = tid >> 5;
    const int wrb  = wave * 16;
    const int rowBlock = blockIdx.x * MT;

    // stage input tile f32 -> bf16 in LDS
    for (int idx = tid; idx < MT * (K0 / 4); idx += 128) {
        int r = idx / (K0 / 4);
        int c = (idx % (K0 / 4)) * 4;
        int gr = rowBlock + r;
        float4 v = make_float4(0.f, 0.f, 0.f, 0.f);
        if (gr < rows) v = *(const float4*)(X + (size_t)gr * inStride + c);
        ushort4 h;
        h.x = f2bf(v.x); h.y = f2bf(v.y); h.z = f2bf(v.z); h.w = f2bf(v.w);
        *(ushort4*)&ldsIn[r * S0 + c] = h;
    }
    __syncthreads();

    floatx8 acc[8];

    // layer 0: K0 -> 128, relu
    gemm_tile<K0, 8>(ldsIn, S0, Wp0, lane, wrb, acc);
    store_tile_lds(ldsMid, S1, wrb, lane, acc, B0, 8, 1);
    __syncthreads();

    if constexpr (NLAYERS == 3) {
        // layer 1: 128 -> 128, relu  (write into ldsIn area, free after barrier)
        gemm_tile<128, 8>(ldsMid, S1, Wp1, lane, wrb, acc);
        store_tile_lds(ldsIn, S1, wrb, lane, acc, B1, 8, 1);
        __syncthreads();
        // layer 2: 128 -> NOUT, finalAct, to global
        gemm_tile<128, NOUT / 16>(ldsIn, S1, Wp2, lane, wrb, acc);
        store_tile_global(Y, outStride, rows, rowBlock, wrb, lane, acc, B2,
                          NOUT / 16, finalAct);
    } else {
        // layer 1: 128 -> NOUT, finalAct, to global
        gemm_tile<128, NOUT / 16>(ldsMid, S1, Wp1, lane, wrb, acc);
        store_tile_global(Y, outStride, rows, rowBlock, wrb, lane, acc, B1,
                          NOUT / 16, finalAct);
    }
}

// ---------------------------------------------------------------------------
// Clause loss: literals are contiguous in groups of 3 per clause, so the
// segment-sum is a 3-element sum. One wave per clause, 4 feats per lane.
//   closs[c,f] = exp(-sum_i softplus(sign_i * q[var_i, f]))
// ---------------------------------------------------------------------------
__device__ __forceinline__ float softplusf(float x) {
    return fmaxf(x, 0.f) + log1pf(__expf(-fabsf(x)));
}

__global__ __launch_bounds__(256) void clause_loss_kernel(
    const float* __restrict__ q, const int* __restrict__ lit_vars,
    const float* __restrict__ lit_sign, float* __restrict__ closs)
{
    int wave = threadIdx.x >> 5, lane = threadIdx.x & 31;
    int c = blockIdx.x * 8 + wave;
    if (c >= NCLAUSES) return;
    int base = c * 3;
    float4 s = make_float4(0.f, 0.f, 0.f, 0.f);
#pragma unroll
    for (int i = 0; i < 3; ++i) {
        int v = lit_vars[base + i];
        float sg = lit_sign[base + i];
        float4 x = *(const float4*)(q + (size_t)v * 128 + lane * 4);
        s.x += softplusf(sg * x.x);
        s.y += softplusf(sg * x.y);
        s.z += softplusf(sg * x.z);
        s.w += softplusf(sg * x.w);
    }
    float4 o;
    o.x = __expf(-s.x); o.y = __expf(-s.y); o.z = __expf(-s.z); o.w = __expf(-s.w);
    *(float4*)(closs + (size_t)c * 128 + lane * 4) = o;
}

// Zero the message columns (128..255) of unit.
__global__ __launch_bounds__(256) void zero_msgs_kernel(float* __restrict__ unit)
{
    int i = blockIdx.x * 256 + threadIdx.x;
    if (i < NVARS * 128) {
        int r = i >> 7, c = i & 127;
        unit[(size_t)r * 256 + 128 + c] = 0.f;
    }
}

// Copy variables_init into unit[:, :128].
__global__ __launch_bounds__(256) void copy_init_kernel(
    const float* __restrict__ vinit, float* __restrict__ unit)
{
    int i = blockIdx.x * 256 + threadIdx.x;
    if (i < NVARS * 128) {
        int r = i >> 7, c = i & 127;
        unit[(size_t)r * 256 + c] = vinit[i];
    }
}

// Scatter clause messages to variables (random lit_vars -> f32 atomics).
// pos literals (sign=+1) add vpos into unit cols 128..191, neg into 192..255.
__global__ __launch_bounds__(256) void scatter_kernel(
    const float* __restrict__ vpos, const float* __restrict__ vneg,
    const int* __restrict__ lit_vars, const float* __restrict__ lit_sign,
    const int* __restrict__ clause_ids, float* __restrict__ unit)
{
    int wave = threadIdx.x >> 5, lane = threadIdx.x & 31;
    int l = blockIdx.x * 8 + wave;
    if (l >= NLITS) return;
    int c = clause_ids[l];
    int v = lit_vars[l];
    float sg = lit_sign[l];
    bool pos = sg > 0.f;
    const float* src = (pos ? vpos : vneg) + (size_t)c * 64;
    float* dst = unit + (size_t)v * 256 + (pos ? 128 : 192);
    atomicAdd(dst + lane * 2,     src[lane * 2]);
    atomicAdd(dst + lane * 2 + 1, src[lane * 2 + 1]);
}

// Gate + LayerNorm: v = (1-fg)*v + fg*nv; LN over 128 feats; writes unit[:, :128].
__global__ __launch_bounds__(256) void gate_ln_kernel(
    float* __restrict__ unit, const float* __restrict__ fg,
    const float* __restrict__ nv, const float* __restrict__ gamma,
    const float* __restrict__ beta)
{
    int wave = threadIdx.x >> 5, lane = threadIdx.x & 31;
    int row = blockIdx.x * 8 + wave;
    if (row >= NVARS) return;
    float* u = unit + (size_t)row * 256;
    float4 v = *(const float4*)(u + lane * 4);
    float4 g = *(const float4*)(fg + (size_t)row * 128 + lane * 4);
    float4 n = *(const float4*)(nv + (size_t)row * 128 + lane * 4);
    float4 w;
    w.x = (1.f - g.x) * v.x + g.x * n.x;
    w.y = (1.f - g.y) * v.y + g.y * n.y;
    w.z = (1.f - g.z) * v.z + g.z * n.z;
    w.w = (1.f - g.w) * v.w + g.w * n.w;
    float s = w.x + w.y + w.z + w.w;
    for (int o = 16; o > 0; o >>= 1) s += __shfl_xor(s, o);
    float mu = s * (1.f / 128.f);
    float dx = w.x - mu, dy = w.y - mu, dz = w.z - mu, dw = w.w - mu;
    float ss = dx * dx + dy * dy + dz * dz + dw * dw;
    for (int o = 16; o > 0; o >>= 1) ss += __shfl_xor(ss, o);
    float rs = rsqrtf(ss * (1.f / 128.f) + 1e-3f);
    float4 ga = *(const float4*)(gamma + lane * 4);
    float4 be = *(const float4*)(beta + lane * 4);
    w.x = dx * rs * ga.x + be.x;
    w.y = dy * rs * ga.y + be.y;
    w.z = dz * rs * ga.z + be.z;
    w.w = dw * rs * ga.w + be.w;
    *(float4*)(u + lane * 4) = w;
}

// Final 128 -> 1 layer: per-row dot with W2 + b2 (wave32 shuffle reduce).
__global__ __launch_bounds__(256) void logits_kernel(
    const float* __restrict__ h, const float* __restrict__ W2,
    const float* __restrict__ b2, float* __restrict__ out)
{
    int wave = threadIdx.x >> 5, lane = threadIdx.x & 31;
    int row = blockIdx.x * 8 + wave;
    if (row >= NVARS) return;
    float4 x = *(const float4*)(h + (size_t)row * 128 + lane * 4);
    float4 w = *(const float4*)(W2 + lane * 4);
    float s = x.x * w.x + x.y * w.y + x.z * w.z + x.w * w.w;
    for (int o = 16; o > 0; o >>= 1) s += __shfl_xor(s, o);
    if (lane == 0) out[row] = s + b2[0];
}

// ---------------------------------------------------------------------------
// Host side
// ---------------------------------------------------------------------------
extern "C" void kernel_launch(void* const* d_in, const int* in_sizes, int n_in,
                              void* d_out, int out_size, void* d_ws, size_t ws_size,
                              hipStream_t stream)
{
    (void)in_sizes; (void)n_in; (void)out_size; (void)ws_size;

    const float* vinit    = (const float*)d_in[0];
    const float* fgW[3]   = {(const float*)d_in[1],  (const float*)d_in[3],  (const float*)d_in[5]};
    const float* fgB[3]   = {(const float*)d_in[2],  (const float*)d_in[4],  (const float*)d_in[6]};
    const float* ln_beta  = (const float*)d_in[7];
    const float* ln_gamma = (const float*)d_in[8];
    const float* qnW[3]   = {(const float*)d_in[9],  (const float*)d_in[11], (const float*)d_in[13]};
    const float* qnB[3]   = {(const float*)d_in[10], (const float*)d_in[12], (const float*)d_in[14]};
    const float* qpW[3]   = {(const float*)d_in[15], (const float*)d_in[17], (const float*)d_in[19]};
    const float* qpB[3]   = {(const float*)d_in[16], (const float*)d_in[18], (const float*)d_in[20]};
    const float* ugW[3]   = {(const float*)d_in[21], (const float*)d_in[23], (const float*)d_in[25]};
    const float* ugB[3]   = {(const float*)d_in[22], (const float*)d_in[24], (const float*)d_in[26]};
    const float* voW[3]   = {(const float*)d_in[27], (const float*)d_in[29], (const float*)d_in[31]};
    const float* voB[3]   = {(const float*)d_in[28], (const float*)d_in[30], (const float*)d_in[32]};
    const float* vqW[3]   = {(const float*)d_in[33], (const float*)d_in[35], (const float*)d_in[37]};
    const float* vqB[3]   = {(const float*)d_in[34], (const float*)d_in[36], (const float*)d_in[38]};
    const int*   lit_vars   = (const int*)d_in[39];
    const float* lit_sign   = (const float*)d_in[40];
    const int*   clause_ids = (const int*)d_in[41];
    float* out = (float*)d_out;

    // workspace layout (bytes)
    char* ws = (char*)d_ws;
    const size_t OFF_UNIT  = 0;                                  // V x 256 f32
    const size_t OFF_Q     = OFF_UNIT  + (size_t)NVARS * 256 * 4; // V x 128 f32 (also h)
    const size_t OFF_CLOSS = OFF_Q     + (size_t)NVARS * 128 * 4; // C x 128 f32
    const size_t OFF_VPOS  = OFF_CLOSS + (size_t)NCLAUSES * 128 * 4;
    const size_t OFF_VNEG  = OFF_VPOS  + (size_t)NCLAUSES * 64 * 4;
    const size_t OFF_FG    = OFF_VNEG  + (size_t)NCLAUSES * 64 * 4;
    const size_t OFF_NV    = OFF_FG    + (size_t)NVARS * 128 * 4;
    const size_t OFF_WP    = OFF_NV    + (size_t)NVARS * 128 * 4;

    float* unit  = (float*)(ws + OFF_UNIT);
    float* q     = (float*)(ws + OFF_Q);
    float* closs = (float*)(ws + OFF_CLOSS);
    float* vpos  = (float*)(ws + OFF_VPOS);
    float* vneg  = (float*)(ws + OFF_VNEG);
    float* fg    = (float*)(ws + OFF_FG);
    float* nv    = (float*)(ws + OFF_NV);

    // packed bf16 weights
    unsigned short* wbase = (unsigned short*)(ws + OFF_WP);
    size_t wo = 0;
    auto walloc = [&](int K, int N) {
        unsigned short* p = wbase + wo; wo += (size_t)K * N; return p;
    };
    unsigned short* pvq[3] = {walloc(128,128), walloc(128,128), walloc(128,128)};
    unsigned short* pqp[3] = {walloc(128,128), walloc(128,128), walloc(128, 64)};
    unsigned short* pqn[3] = {walloc(128,128), walloc(128,128), walloc(128, 64)};
    unsigned short* pug[3] = {walloc(256,128), walloc(128,128), walloc(128,128)};
    unsigned short* pfg[3] = {walloc(256,128), walloc(128,128), walloc(128,128)};
    unsigned short* pvo[2] = {walloc(128,128), walloc(128,128)};

    auto pack = [&](const float* W, unsigned short* dst, int K, int N) {
        int total = K * N;
        pack_weights<<<(total + 255) / 256, 256, 0, stream>>>(W, dst, K, N);
    };
    for (int i = 0; i < 3; ++i) pack(vqW[i], pvq[i], 128, 128);
    pack(qpW[0], pqp[0], 128, 128); pack(qpW[1], pqp[1], 128, 128); pack(qpW[2], pqp[2], 128, 64);
    pack(qnW[0], pqn[0], 128, 128); pack(qnW[1], pqn[1], 128, 128); pack(qnW[2], pqn[2], 128, 64);
    pack(ugW[0], pug[0], 256, 128); pack(ugW[1], pug[1], 128, 128); pack(ugW[2], pug[2], 128, 128);
    pack(fgW[0], pfg[0], 256, 128); pack(fgW[1], pfg[1], 128, 128); pack(fgW[2], pfg[2], 128, 128);
    pack(voW[0], pvo[0], 128, 128); pack(voW[1], pvo[1], 128, 128);

    const int VB = (NVARS + 63) / 64;      // row-tile blocks for variable MLPs
    const int CB = (NCLAUSES + 63) / 64;   // row-tile blocks for clause MLPs
    const int VE = (NVARS * 128 + 255) / 256;

    copy_init_kernel<<<VE, 256, 0, stream>>>(vinit, unit);

    for (int r = 0; r < NROUNDS; ++r) {
        // 1. variables_query: unit[:, :128] -> q
        mlp_wmma<128, 128, 3><<<VB, 128, 0, stream>>>(
            unit, 256, NVARS, pvq[0], vqB[0], pvq[1], vqB[1], pvq[2], vqB[2],
            q, 128, 0);

        // 2. softplus/segment-sum/exp per clause
        clause_loss_kernel<<<(NCLAUSES + 7) / 8, 256, 0, stream>>>(
            q, lit_vars, lit_sign, closs);

        // 3. clause message MLPs (dominant GEMMs, 84000 rows)
        mlp_wmma<128, 64, 3><<<CB, 128, 0, stream>>>(
            closs, 128, NCLAUSES, pqp[0], qpB[0], pqp[1], qpB[1], pqp[2], qpB[2],
            vpos, 64, 0);
        mlp_wmma<128, 64, 3><<<CB, 128, 0, stream>>>(
            closs, 128, NCLAUSES, pqn[0], qnB[0], pqn[1], qnB[1], pqn[2], qnB[2],
            vneg, 64, 0);

        // 4. scatter messages to variables (into unit cols 128..255)
        zero_msgs_kernel<<<VE, 256, 0, stream>>>(unit);
        scatter_kernel<<<(NLITS + 7) / 8, 256, 0, stream>>>(
            vpos, vneg, lit_vars, lit_sign, clause_ids, unit);

        // 5. gates on concat input [V,256]
        mlp_wmma<256, 128, 3><<<VB, 128, 0, stream>>>(
            unit, 256, NVARS, pfg[0], fgB[0], pfg[1], fgB[1], pfg[2], fgB[2],
            fg, 128, 2 /*sigmoid*/);
        mlp_wmma<256, 128, 3><<<VB, 128, 0, stream>>>(
            unit, 256, NVARS, pug[0], ugB[0], pug[1], ugB[1], pug[2], ugB[2],
            nv, 128, 0);

        // 6. gate + LayerNorm, writes new state into unit[:, :128]
        gate_ln_kernel<<<(NVARS + 7) / 8, 256, 0, stream>>>(
            unit, fg, nv, ln_gamma, ln_beta);

        // 7. variables_output: two WMMA layers into q, then 128->1 dot
        mlp_wmma<128, 128, 2><<<VB, 128, 0, stream>>>(
            unit, 256, NVARS, pvo[0], voB[0], pvo[1], voB[1], nullptr, nullptr,
            q, 128, 1 /*relu*/);
        logits_kernel<<<(NVARS + 7) / 8, 256, 0, stream>>>(
            q, voW[2], voB[2], out + (size_t)r * NVARS);
    }
}
